// EdgeUpdateModule_50199577756223
// MI455X (gfx1250) — matrix operand order repaired and checked by reading, not compiled, not run on hardware
//
#include <hip/hip_runtime.h>
#include <hip/hip_bf16.h>

typedef __attribute__((ext_vector_type(16))) _Float16 v16h;
typedef __attribute__((ext_vector_type(8)))  _Float16 v8h;
typedef __attribute__((ext_vector_type(8)))  float    v8f;

#define NODE_DIM 128
#define EDGE_DIM 64
#define HIDDEN   128
#define IN_DIM   320
#define N_TYPES  4
#define LN_EPS   1e-5f

// ---------------- workspace layout (ints) ----------------
// ws[0..3]   counts per type
// ws[4..7]   scatter cursors
// ws[8..12]  tile bases (tiles of 16), ws[12] = total tiles
// ws[32..]   perm[E + 64]  (type-sorted edge ids, -1 = padding)
// after perm: f16 swizzled W1 (4*320*128 halves), then W2 (4*128*64 halves)

__global__ void k_init(int* ws) {
    if (threadIdx.x < 8) ws[threadIdx.x] = 0;
}

__global__ void k_count(const int* __restrict__ etype, int* __restrict__ counts, int E) {
    int i = blockIdx.x * blockDim.x + threadIdx.x;
    if (i < E) atomicAdd(&counts[etype[i] & 3], 1);
}

__global__ void k_scan(int* ws, int E) {
    int* counts  = ws + 0;
    int* cursors = ws + 4;
    int* meta    = ws + 8;
    int* perm    = ws + 32;
    int base = 0;
    for (int t = 0; t < N_TYPES; t++) {
        meta[t]    = base >> 4;      // tile base for type t
        cursors[t] = base;
        int c   = counts[t];
        int pad = (c + 15) & ~15;
        for (int i = c; i < pad; i++) perm[base + i] = -1;  // <=15 sentinels/type
        base += pad;
    }
    meta[4] = base >> 4;             // total tiles
}

__global__ void k_scatter(const int* __restrict__ etype, int* __restrict__ cursors,
                          int* __restrict__ perm, int E) {
    int i = blockIdx.x * blockDim.x + threadIdx.x;
    if (i < E) {
        int t = etype[i] & 3;
        int p = atomicAdd(&cursors[t], 1);
        perm[p] = i;
    }
}

// Convert W1/W2 f32 -> f16 and pre-swizzle into WMMA B-fragment-major layout:
// B 32x16 f16 fragment: lane = (n%16) + 16*kh, element j = k%16, kh = (k%32)/16.
// Fragment blocks stored contiguously per (ntile, kstep): [lane][16 halves].
__global__ void k_swizzle(const float* __restrict__ W1, const float* __restrict__ W2,
                          _Float16* __restrict__ w1o, _Float16* __restrict__ w2o) {
    const int n1 = N_TYPES * IN_DIM * HIDDEN;   // 163840
    const int n2 = N_TYPES * HIDDEN * EDGE_DIM; // 32768
    for (int idx = blockIdx.x * blockDim.x + threadIdx.x; idx < n1 + n2;
         idx += gridDim.x * blockDim.x) {
        if (idx < n1) {
            int t = idx / (IN_DIM * HIDDEN);
            int r = idx % (IN_DIM * HIDDEN);
            int k = r >> 7, n = r & 127;
            int ks = k >> 5, kh = (k >> 4) & 1, j = k & 15;
            int nt = n >> 4, lane = (n & 15) + (kh << 4);
            w1o[t * (IN_DIM * HIDDEN) + (((nt * 10 + ks) * 32 + lane) << 4) + j] =
                (_Float16)W1[idx];
        } else {
            int i2 = idx - n1;
            int t = i2 / (HIDDEN * EDGE_DIM);
            int r = i2 % (HIDDEN * EDGE_DIM);
            int k = r >> 6, n = r & 63;
            int ks = k >> 5, kh = (k >> 4) & 1, j = k & 15;
            int nt = n >> 4, lane = (n & 15) + (kh << 4);
            w2o[t * (HIDDEN * EDGE_DIM) + (((nt * 4 + ks) * 32 + lane) << 4) + j] =
                (_Float16)W2[i2];
        }
    }
}

// A 16x32 f16 fragment: lane = m + 16*kh; element j -> K = 16*kh + j.
// Load as two 16B chunks from row-major LDS.
__device__ __forceinline__ v16h load_afrag(const _Float16* rowbase, int rowstride,
                                           int m, int kh, int kbase) {
    const _Float16* ap = rowbase + m * rowstride + kbase + (kh << 4);
    v8h lo = *(const v8h*)ap;
    v8h hi = *(const v8h*)(ap + 8);
    v16h a;
#pragma unroll
    for (int i = 0; i < 8; i++) { a[i] = lo[i]; a[8 + i] = hi[i]; }
    return a;
}

__global__ __launch_bounds__(128) void k_compute(
    const float* __restrict__ x, const float* __restrict__ ea,
    const float* __restrict__ b1, const float* __restrict__ lnw,
    const float* __restrict__ lnb, const float* __restrict__ b2,
    const _Float16* __restrict__ w1, const _Float16* __restrict__ w2,
    const int* __restrict__ ei, const int* __restrict__ meta,
    const int* __restrict__ perm, float* __restrict__ out, int E)
{
    __shared__ _Float16 atile[16 * IN_DIM];   // 10 KB
    __shared__ float    hbuf[16 * HIDDEN];    //  8 KB
    __shared__ _Float16 a2[16 * HIDDEN];      //  4 KB
    __shared__ float    outs[16 * EDGE_DIM];  //  4 KB
    __shared__ float    sb1[HIDDEN], slw[HIDDEN], slb[HIDDEN], sb2[EDGE_DIM];
    __shared__ float    ps1[16 * 8], ps2[16 * 8], smu[16], srs[16];

    const int tid  = threadIdx.x;
    const int lane = tid & 31;
    const int w    = tid >> 5;                 // wave id 0..3

    const int tb1 = meta[1], tb2 = meta[2], tb3 = meta[3], total = meta[4];
    int curType = -1;
    const _Float16* w1t = w1;
    const _Float16* w2t = w2;

    for (int tile = blockIdx.x; tile < total; tile += gridDim.x) {
        int t = (tile >= tb3) ? 3 : (tile >= tb2) ? 2 : (tile >= tb1) ? 1 : 0;
        if (t != curType) {
            curType = t;
            w1t = w1 + t * (IN_DIM * HIDDEN);
            w2t = w2 + t * (HIDDEN * EDGE_DIM);
            __syncthreads();
            sb1[tid] = b1[t * HIDDEN + tid];
            slw[tid] = lnw[t * HIDDEN + tid];
            slb[tid] = lnb[t * HIDDEN + tid];
            if (tid < EDGE_DIM) sb2[tid] = b2[t * EDGE_DIM + tid];
            __syncthreads();
        }

        // ---- gather A tile: [16 edges][320] f32 -> f16 in LDS ----
        {
            int row = tid >> 3, q = tid & 7;               // 8 threads per row
            int e = perm[(tile << 4) + row];
            const float4* sx = (e >= 0) ? (const float4*)(x + (size_t)ei[e] * NODE_DIM) : nullptr;
            const float4* dx = (e >= 0) ? (const float4*)(x + (size_t)ei[E + e] * NODE_DIM) : nullptr;
            const float4* ex = (e >= 0) ? (const float4*)(ea + (size_t)e * EDGE_DIM) : nullptr;
#pragma unroll
            for (int i = 0; i < 10; i++) {
                int c4 = q + (i << 3);                     // 0..79 float4s
                float4 v;
                if (e < 0)        v = make_float4(0.f, 0.f, 0.f, 0.f);
                else if (c4 < 32) v = sx[c4];
                else if (c4 < 64) v = dx[c4 - 32];
                else              v = ex[c4 - 64];
                _Float16* p = &atile[row * IN_DIM + (c4 << 2)];
                p[0] = (_Float16)v.x; p[1] = (_Float16)v.y;
                p[2] = (_Float16)v.z; p[3] = (_Float16)v.w;
            }
        }
        __syncthreads();

        // ---- GEMM1: [16x320] x [320x128] -> h fp32, per wave 2 N-tiles ----
        {
            v8f c0 = {}; v8f c1 = {};
            const int m = lane & 15, kh = lane >> 4;
            const int nt0 = w * 2, nt1 = w * 2 + 1;
#pragma unroll
            for (int ks = 0; ks < 10; ks++) {
                v16h a = load_afrag(atile, IN_DIM, m, kh, ks << 5);
                v16h bA = *(const v16h*)&w1t[((nt0 * 10 + ks) * 32 + lane) << 4];
                v16h bB = *(const v16h*)&w1t[((nt1 * 10 + ks) * 32 + lane) << 4];
                c0 = __builtin_amdgcn_wmma_f32_16x16x32_f16(false, a, false, bA,
                                                            (short)0, c0, false, false);
                c1 = __builtin_amdgcn_wmma_f32_16x16x32_f16(false, a, false, bB,
                                                            (short)0, c1, false, false);
            }
            const int n = lane & 15, mh = lane >> 4;
#pragma unroll
            for (int r = 0; r < 8; r++) {
                hbuf[(r + (mh << 3)) * HIDDEN + (nt0 << 4) + n] = c0[r];
                hbuf[(r + (mh << 3)) * HIDDEN + (nt1 << 4) + n] = c1[r];
            }
        }
        __syncthreads();

        // ---- LayerNorm over 128 (with +b1), two-stage reduction ----
        {
            int row = tid & 15, seg = tid >> 4;            // 16 rows x 8 segs
            float s1 = 0.f, s2 = 0.f;
#pragma unroll
            for (int i = 0; i < 16; i++) {
                int c = (seg << 4) + i;
                float v = hbuf[row * HIDDEN + c] + sb1[c];
                hbuf[row * HIDDEN + c] = v;
                s1 += v; s2 += v * v;
            }
            ps1[row * 8 + seg] = s1; ps2[row * 8 + seg] = s2;
        }
        __syncthreads();
        if (tid < 16) {
            float s1 = 0.f, s2 = 0.f;
#pragma unroll
            for (int i = 0; i < 8; i++) { s1 += ps1[tid * 8 + i]; s2 += ps2[tid * 8 + i]; }
            float mu  = s1 * (1.f / HIDDEN);
            float var = s2 * (1.f / HIDDEN) - mu * mu;
            smu[tid] = mu;
            srs[tid] = rsqrtf(var + LN_EPS);
        }
        __syncthreads();
        {   // apply LN affine + exact GELU, emit f16 A2
            int row = tid >> 3, c0i = (tid & 7) << 4;
            float mu = smu[row], rs = srs[row];
#pragma unroll
            for (int i = 0; i < 16; i++) {
                int c = c0i + i;
                float v = (hbuf[row * HIDDEN + c] - mu) * rs * slw[c] + slb[c];
                float g = 0.5f * v * (1.f + erff(v * 0.70710678118f));
                a2[row * HIDDEN + c] = (_Float16)g;
            }
        }
        __syncthreads();

        // ---- GEMM2: [16x128] x [128x64] -> out, per wave 1 N-tile ----
        {
            v8f acc = {};
            const int m = lane & 15, kh = lane >> 4;
#pragma unroll
            for (int ks = 0; ks < 4; ks++) {
                v16h a  = load_afrag(a2, HIDDEN, m, kh, ks << 5);
                v16h bv = *(const v16h*)&w2t[((w * 4 + ks) * 32 + lane) << 4];
                acc = __builtin_amdgcn_wmma_f32_16x16x32_f16(false, a, false, bv,
                                                             (short)0, acc, false, false);
            }
            const int n = lane & 15, mh = lane >> 4;
#pragma unroll
            for (int r = 0; r < 8; r++) {
                int col = (w << 4) + n;
                outs[(r + (mh << 3)) * EDGE_DIM + col] = acc[r] + sb2[col];
            }
        }
        __syncthreads();

        // ---- coalesced scatter of 16 output rows ----
        {
            int row = tid >> 3, q = tid & 7;
            int e = perm[(tile << 4) + row];
            if (e >= 0) {
                float4 v0 = *(float4*)&outs[row * EDGE_DIM + (q << 3)];
                float4 v1 = *(float4*)&outs[row * EDGE_DIM + (q << 3) + 4];
                float4* op = (float4*)(out + (size_t)e * EDGE_DIM + (q << 3));
                op[0] = v0; op[1] = v1;
            }
        }
        __syncthreads();
    }
}

extern "C" void kernel_launch(void* const* d_in, const int* in_sizes, int n_in,
                              void* d_out, int out_size, void* d_ws, size_t ws_size,
                              hipStream_t stream) {
    const float* x   = (const float*)d_in[0];
    const float* ea  = (const float*)d_in[1];
    const float* W1  = (const float*)d_in[2];
    const float* b1  = (const float*)d_in[3];
    const float* lnw = (const float*)d_in[4];
    const float* lnb = (const float*)d_in[5];
    const float* W2  = (const float*)d_in[6];
    const float* b2  = (const float*)d_in[7];
    const int*   ei  = (const int*)d_in[8];
    const int*   et  = (const int*)d_in[9];
    float*       out = (float*)d_out;

    const int E = in_sizes[1] / EDGE_DIM;   // 800000

    int* ws       = (int*)d_ws;
    int* counts   = ws + 0;
    int* cursors  = ws + 4;
    int* meta     = ws + 8;
    int* perm     = ws + 32;
    _Float16* w1h = (_Float16*)(ws + 32 + E + 64);        // 32B-aligned
    _Float16* w2h = w1h + N_TYPES * IN_DIM * HIDDEN;

    k_init<<<1, 32, 0, stream>>>(ws);
    k_count<<<(E + 255) / 256, 256, 0, stream>>>(et, counts, E);
    k_scan<<<1, 1, 0, stream>>>(ws, E);
    k_scatter<<<(E + 255) / 256, 256, 0, stream>>>(et, cursors, perm, E);
    k_swizzle<<<512, 256, 0, stream>>>(W1, W2, w1h, w2h);
    k_compute<<<2048, 128, 0, stream>>>(x, ea, b1, lnw, lnb, b2,
                                        w1h, w2h, ei, meta, perm, out, E);
}
